// EGNN_16862041604107
// MI455X (gfx1250) — matrix operand order repaired and checked by reading, not compiled, not run on hardware
//
#include <hip/hip_runtime.h>

// ---------------------------------------------------------------------------
// EGNN forward for MI455X (gfx1250): bf16 WMMA (V_WMMA_F32_16X16X32_BF16) for
// all H=128 GEMMs, fp32 LayerNorm/SiLU/atomics. wave32; one wave = 16-row tile.
// ---------------------------------------------------------------------------

typedef __bf16 bf16_t;
typedef __bf16 v16bf __attribute__((ext_vector_type(16)));
typedef float  v8f   __attribute__((ext_vector_type(8)));
typedef __bf16 bf16x8 __attribute__((ext_vector_type(8)));

#define HF     128
#define DIN    261      // 2H + 1 + A
#define KPAD1  288      // DIN padded to 9*32
#define KS1    9
#define LAYERS 4

__device__ __forceinline__ void lds_fence() {
  asm volatile("s_wait_dscnt 0x0" ::: "memory");
}

// A fragment, 16-bit, 16x32 (MxK). lane l: m = l&15.
// elems 0..15 -> k = ks*32 + (l>=16?16:0) + e
// caller passes p = &act[m][ks*32 + (l>=16?16:0)]
__device__ __forceinline__ v16bf load_a16(const bf16_t* p) {
  bf16x8 lo = *reinterpret_cast<const bf16x8*>(p);
  bf16x8 hi = *reinterpret_cast<const bf16x8*>(p + 8);
  v16bf a;
#pragma unroll
  for (int i = 0; i < 8; ++i) { a[i] = lo[i]; a[i + 8] = hi[i]; }
  return a;
}

// B fragment, 16-bit, 32x16 (KxN). lane l: n = l&15, k = (l>=16?16:0)+e.
// Weights are pre-swizzled so each lane's 16 values are contiguous.
__device__ __forceinline__ v16bf load_b16(const bf16_t* p) {
  bf16x8 lo = *reinterpret_cast<const bf16x8*>(p);
  bf16x8 hi = *reinterpret_cast<const bf16x8*>(p + 8);
  v16bf b;
#pragma unroll
  for (int i = 0; i < 8; ++i) { b[i] = lo[i]; b[i + 8] = hi[i]; }
  return b;
}

// One wave computes a [16 x 128] = act[16 x 32*KS] @ W tile via WMMA.
// D layout: lane l holds rows m=(l>=16?8:0)+v (v=0..7), col n = 16*nt + (l&15).
template <int KS, int STRIDE>
__device__ __forceinline__ void wave_gemm(const bf16_t* act, const bf16_t* wf,
                                          int lane, float z[8][8]) {
  const int nl  = lane & 15;
  const int hb8 = (lane & 16) ? 16 : 0;
  const bf16_t* arow = act + nl * STRIDE + hb8;  // A row m = nl
#pragma unroll
  for (int nt = 0; nt < 8; ++nt) {
    v8f acc = {};
#pragma unroll
    for (int ks = 0; ks < KS; ++ks) {
      v16bf a = load_a16(arow + ks * 32);
      v16bf b = load_b16(wf + (nt * KS + ks) * 512 + lane * 16);
      acc = __builtin_amdgcn_wmma_f32_16x16x32_bf16(
          false, a, false, b, (short)0, acc, false, false);
    }
#pragma unroll
    for (int v = 0; v < 8; ++v) z[nt][v] = acc[v];
  }
}

// bias + LayerNorm(128) + SiLU, in place. Row stats reduced over the 16 lanes
// of each half-wave (they jointly own a row's 128 columns) via shfl_xor.
__device__ __forceinline__ void ln_silu(float z[8][8], const float* bias,
                                        const float* g, const float* b,
                                        int lane) {
  const int nl = lane & 15;
  float rs[8], rq[8];
#pragma unroll
  for (int v = 0; v < 8; ++v) { rs[v] = 0.f; rq[v] = 0.f; }
#pragma unroll
  for (int nt = 0; nt < 8; ++nt) {
    float bv = bias[nt * 16 + nl];
#pragma unroll
    for (int v = 0; v < 8; ++v) {
      float t = z[nt][v] + bv;
      z[nt][v] = t; rs[v] += t; rq[v] += t * t;
    }
  }
#pragma unroll
  for (int mask = 1; mask < 16; mask <<= 1) {
#pragma unroll
    for (int v = 0; v < 8; ++v) {
      rs[v] += __shfl_xor(rs[v], mask, 32);
      rq[v] += __shfl_xor(rq[v], mask, 32);
    }
  }
  float mean[8], rstd[8];
#pragma unroll
  for (int v = 0; v < 8; ++v) {
    mean[v] = rs[v] * (1.f / 128.f);
    float var = rq[v] * (1.f / 128.f) - mean[v] * mean[v];
    rstd[v] = rsqrtf(var + 1e-5f);
  }
#pragma unroll
  for (int nt = 0; nt < 8; ++nt) {
    float gv = g[nt * 16 + nl], bv = b[nt * 16 + nl];
#pragma unroll
    for (int v = 0; v < 8; ++v) {
      float t = (z[nt][v] - mean[v]) * rstd[v] * gv + bv;
      z[nt][v] = t / (1.f + __expf(-t));  // silu
    }
  }
}

template <int STRIDE>
__device__ __forceinline__ void store_act(bf16_t* dst, const float z[8][8],
                                          int lane) {
  const int nl = lane & 15;
  const int hb8 = (lane & 16) ? 8 : 0;
#pragma unroll
  for (int nt = 0; nt < 8; ++nt)
#pragma unroll
    for (int v = 0; v < 8; ++v)
      dst[(hb8 + v) * STRIDE + nt * 16 + nl] = (bf16_t)z[nt][v];
}

// --------------------------- small helper kernels ---------------------------

__global__ void egnn_degree(const int* __restrict__ src, float* __restrict__ c,
                            int E) {
  int e = blockIdx.x * blockDim.x + threadIdx.x;
  if (e < E) atomicAdd(&c[src[e]], 1.0f);
}

__global__ void egnn_emb(const float* __restrict__ hin,
                         const float* __restrict__ w,
                         const float* __restrict__ b, float* __restrict__ h32,
                         bf16_t* __restrict__ hbf, int N) {
  int idx = blockIdx.x * blockDim.x + threadIdx.x;
  if (idx >= N * HF) return;
  int node = idx >> 7, o = idx & 127;
  float s = b[o];
#pragma unroll
  for (int k = 0; k < 16; ++k) s += hin[node * 16 + k] * w[k * HF + o];
  h32[idx] = s;
  hbf[idx] = (bf16_t)s;
}

// fp32 [Kreal x 128] row-major -> bf16 WMMA B fragments (fragment-linear):
// out[((nt*KS+ks)*32 + lane)*16 + e] = W[ks*32 + (lane>=16?16:0)+e][nt*16+(lane&15)]
__global__ void egnn_wfrag(const float* __restrict__ W, bf16_t* __restrict__ out,
                           int Kreal, int Kpad) {
  int idx = blockIdx.x * blockDim.x + threadIdx.x;
  int KS = Kpad >> 5;
  int total = KS * 8 * 512;
  if (idx >= total) return;
  int e = idx & 15, lane = (idx >> 4) & 31, fi = idx >> 9;
  int nt = fi / KS, ks = fi - nt * KS;
  int n = nt * 16 + (lane & 15);
  int k = ks * 32 + ((lane >= 16) ? 16 : 0) + e;
  float v = (k < Kreal) ? W[k * HF + n] : 0.f;
  out[idx] = (bf16_t)v;
}

__global__ void egnn_final(const float* __restrict__ h32,
                           const float* __restrict__ xw,
                           const float* __restrict__ ow,
                           const float* __restrict__ ob,
                           float* __restrict__ out, int N) {
  int idx = blockIdx.x * blockDim.x + threadIdx.x;
  if (idx < N * 3) { out[idx] = xw[idx]; return; }
  int r = idx - N * 3;
  if (r >= N * 16) return;
  int node = r >> 4, o = r & 15;
  float s = ob[o];
#pragma unroll 8
  for (int k = 0; k < HF; ++k) s += h32[node * HF + k] * ow[k * 16 + o];
  out[idx] = s;
}

// ------------------------------- edge kernel --------------------------------
// 8 waves/block, each wave owns one 16-edge tile. Three WMMA MLP stages.

__global__ __launch_bounds__(256) void egnn_edge(
    const int* __restrict__ src, const int* __restrict__ dst,
    const float* __restrict__ xw, const float* __restrict__ eattr,
    const bf16_t* __restrict__ hbf,
    const bf16_t* __restrict__ w1f, const float* __restrict__ b1,
    const float* __restrict__ g1, const float* __restrict__ bb1,
    const bf16_t* __restrict__ w2f, const float* __restrict__ b2,
    const float* __restrict__ g2, const float* __restrict__ bb2,
    const bf16_t* __restrict__ w3f, const float* __restrict__ b3,
    const float* __restrict__ g3, const float* __restrict__ bb3,
    const float* __restrict__ w4, const float* __restrict__ b4,
    float* __restrict__ aggh, float* __restrict__ aggx, int E) {
  __shared__ __align__(16) bf16_t sA[8][16][KPAD1];
  __shared__ float sD[8][16][4];
  __shared__ int sI[8][16];

  const int wid = threadIdx.x >> 5;
  const int lane = threadIdx.x & 31;
  const int tile = blockIdx.x * 8 + wid;
  const int base = tile * 16;
  if (base >= E) return;

  const int m = lane & 15, half = lane >> 4;
  const int nl = lane & 15;
  const int hb8 = (lane & 16) ? 8 : 0;

  // ---- stage m_in = [h_i | h_j | d2 | attr | 0-pad] as bf16 in LDS ----
  {
    int e = base + m;
    int jn = src[e], in_ = dst[e];
    const bf16x8* pi =
        reinterpret_cast<const bf16x8*>(hbf + (size_t)in_ * HF + half * 64);
    bf16x8* qi = reinterpret_cast<bf16x8*>(&sA[wid][m][half * 64]);
#pragma unroll
    for (int t = 0; t < 8; ++t) qi[t] = pi[t];
    const bf16x8* pj =
        reinterpret_cast<const bf16x8*>(hbf + (size_t)jn * HF + half * 64);
    bf16x8* qj = reinterpret_cast<bf16x8*>(&sA[wid][m][HF + half * 64]);
#pragma unroll
    for (int t = 0; t < 8; ++t) qj[t] = pj[t];
    if (half == 0) {
      float dx = xw[in_ * 3 + 0] - xw[jn * 3 + 0];
      float dy = xw[in_ * 3 + 1] - xw[jn * 3 + 1];
      float dz = xw[in_ * 3 + 2] - xw[jn * 3 + 2];
      sD[wid][m][0] = dx; sD[wid][m][1] = dy; sD[wid][m][2] = dz;
      sI[wid][m] = in_;
      sA[wid][m][256] = (bf16_t)(dx * dx + dy * dy + dz * dz);
#pragma unroll
      for (int a = 0; a < 4; ++a)
        sA[wid][m][257 + a] = (bf16_t)eattr[(size_t)e * 4 + a];
#pragma unroll
      for (int zc = DIN; zc < KPAD1; ++zc) sA[wid][m][zc] = (bf16_t)0.f;
    }
  }
  lds_fence();

  float z[8][8];
  // ---- MLP1: [16x288] @ pe_w1 -> LN -> SiLU ----
  wave_gemm<KS1, KPAD1>(&sA[wid][0][0], w1f, lane, z);
  ln_silu(z, b1, g1, bb1, lane);
  lds_fence();
  store_act<KPAD1>(&sA[wid][0][0], z, lane);
  lds_fence();
  // ---- MLP2: -> mh ----
  wave_gemm<4, KPAD1>(&sA[wid][0][0], w2f, lane, z);
  ln_silu(z, b2, g2, bb2, lane);
  // segment-sum mh at target nodes
#pragma unroll
  for (int nt = 0; nt < 8; ++nt) {
    int n = nt * 16 + nl;
#pragma unroll
    for (int v = 0; v < 8; ++v) {
      int node = sI[wid][hb8 + v];
      atomicAdd(&aggh[(size_t)node * HF + n], z[nt][v]);
    }
  }
  lds_fence();
  store_act<KPAD1>(&sA[wid][0][HF], z, lane);
  lds_fence();
  // ---- MLP3: -> t2; then scalar gate = t2 @ px_w2 + b ----
  wave_gemm<4, KPAD1>(&sA[wid][0][HF], w3f, lane, z);
  ln_silu(z, b3, g3, bb3, lane);
  float pv[8];
#pragma unroll
  for (int v = 0; v < 8; ++v) pv[v] = 0.f;
#pragma unroll
  for (int nt = 0; nt < 8; ++nt) {
    float wv = w4[nt * 16 + nl];
#pragma unroll
    for (int v = 0; v < 8; ++v) pv[v] += z[nt][v] * wv;
  }
#pragma unroll
  for (int mask = 1; mask < 16; mask <<= 1)
#pragma unroll
    for (int v = 0; v < 8; ++v) pv[v] += __shfl_xor(pv[v], mask, 32);
  float bsc = b4[0];
  if (nl < 3) {
#pragma unroll
    for (int v = 0; v < 8; ++v) {
      int mm = hb8 + v;
      int node = sI[wid][mm];
      atomicAdd(&aggx[(size_t)node * 3 + nl], sD[wid][mm][nl] * (pv[v] + bsc));
    }
  }
}

// ------------------------------- node kernel --------------------------------

__global__ __launch_bounds__(256) void egnn_node(
    const float* __restrict__ aggh, const float* __restrict__ aggx,
    const float* __restrict__ c, float* __restrict__ h32,
    bf16_t* __restrict__ hbf, float* __restrict__ xw,
    const bf16_t* __restrict__ w1f, const float* __restrict__ b1,
    const float* __restrict__ g1, const float* __restrict__ bb1,
    const bf16_t* __restrict__ w2f, const float* __restrict__ b2, int N) {
  __shared__ __align__(16) bf16_t sU[8][16][256];

  const int wid = threadIdx.x >> 5;
  const int lane = threadIdx.x & 31;
  const int tile = blockIdx.x * 8 + wid;
  const int base = tile * 16;
  if (base >= N) return;

  const int m = lane & 15, half = lane >> 4;
  const int nl = lane & 15;
  const int hb8 = (lane & 16) ? 8 : 0;

  // ---- stage u = [h | agg_h] as bf16 ----
  {
    int node = base + m;
    const float4* ph =
        reinterpret_cast<const float4*>(h32 + (size_t)node * HF + half * 64);
    const float4* pa =
        reinterpret_cast<const float4*>(aggh + (size_t)node * HF + half * 64);
#pragma unroll
    for (int t = 0; t < 16; ++t) {
      float4 f = ph[t];
      int col = half * 64 + t * 4;
      sU[wid][m][col + 0] = (bf16_t)f.x; sU[wid][m][col + 1] = (bf16_t)f.y;
      sU[wid][m][col + 2] = (bf16_t)f.z; sU[wid][m][col + 3] = (bf16_t)f.w;
    }
#pragma unroll
    for (int t = 0; t < 16; ++t) {
      float4 f = pa[t];
      int col = HF + half * 64 + t * 4;
      sU[wid][m][col + 0] = (bf16_t)f.x; sU[wid][m][col + 1] = (bf16_t)f.y;
      sU[wid][m][col + 2] = (bf16_t)f.z; sU[wid][m][col + 3] = (bf16_t)f.w;
    }
  }
  lds_fence();

  float z[8][8];
  // ---- t3 = silu(ln(u @ ph_w1 + b1)) ----
  wave_gemm<8, 256>(&sU[wid][0][0], w1f, lane, z);
  ln_silu(z, b1, g1, bb1, lane);
  lds_fence();
  store_act<256>(&sU[wid][0][0], z, lane);
  lds_fence();
  // ---- h = t3 @ ph_w2 + b2 + h ----
  wave_gemm<4, 256>(&sU[wid][0][0], w2f, lane, z);
#pragma unroll
  for (int nt = 0; nt < 8; ++nt) {
    int n = nt * 16 + nl;
    float bv = b2[n];
#pragma unroll
    for (int v = 0; v < 8; ++v) {
      int node = base + hb8 + v;
      size_t k = (size_t)node * HF + n;
      float val = z[nt][v] + bv + h32[k];
      h32[k] = val;
      hbf[k] = (bf16_t)val;
    }
  }
  // ---- x += agg_x / c ----
  if (nl < 3) {
#pragma unroll
    for (int v = 0; v < 8; ++v) {
      int node = base + hb8 + v;
      size_t k = (size_t)node * 3 + nl;
      xw[k] += aggx[k] / c[node];
    }
  }
}

// ------------------------------- launcher -----------------------------------

extern "C" void kernel_launch(void* const* d_in, const int* in_sizes, int n_in,
                              void* d_out, int out_size, void* d_ws,
                              size_t ws_size, hipStream_t stream) {
  const float* x_in  = (const float*)d_in[0];
  const float* h_in  = (const float*)d_in[1];
  const int*   edges = (const int*)d_in[2];
  const float* eattr = (const float*)d_in[3];
  const float* emb_w = (const float*)d_in[4];
  const float* emb_b = (const float*)d_in[5];
  const float* out_w = (const float*)d_in[6];
  const float* out_b = (const float*)d_in[7];
  const float* pe_w1 = (const float*)d_in[8];
  const float* pe_b1 = (const float*)d_in[9];
  const float* pe_g1 = (const float*)d_in[10];
  const float* pe_bb1= (const float*)d_in[11];
  const float* pe_w2 = (const float*)d_in[12];
  const float* pe_b2 = (const float*)d_in[13];
  const float* pe_g2 = (const float*)d_in[14];
  const float* pe_bb2= (const float*)d_in[15];
  const float* px_w1 = (const float*)d_in[16];
  const float* px_b1 = (const float*)d_in[17];
  const float* px_g1 = (const float*)d_in[18];
  const float* px_bb1= (const float*)d_in[19];
  const float* px_w2 = (const float*)d_in[20];
  const float* px_b2 = (const float*)d_in[21];
  const float* ph_w1 = (const float*)d_in[22];
  const float* ph_b1 = (const float*)d_in[23];
  const float* ph_g1 = (const float*)d_in[24];
  const float* ph_bb1= (const float*)d_in[25];
  const float* ph_w2 = (const float*)d_in[26];
  const float* ph_b2 = (const float*)d_in[27];

  const int N = in_sizes[0] / 3;
  const int E = in_sizes[2] / 2;
  const int* src = edges;
  const int* dst = edges + E;

  // ---- workspace carve (all 256B aligned) ----
  char* p = (char*)d_ws;
  auto carve = [&](size_t bytes) -> char* {
    char* r = p;
    p += (bytes + 255) & ~(size_t)255;
    return r;
  };
  float*  h32   = (float*)carve((size_t)N * HF * 4);
  bf16_t* hbf   = (bf16_t*)carve((size_t)N * HF * 2);
  float*  aggh  = (float*)carve((size_t)N * HF * 4);
  float*  aggx  = (float*)carve((size_t)N * 3 * 4);
  float*  cdeg  = (float*)carve((size_t)N * 4);
  float*  xw    = (float*)carve((size_t)N * 3 * 4);
  bf16_t* f_pe1 = (bf16_t*)carve((size_t)LAYERS * KPAD1 * HF * 2);
  bf16_t* f_pe2 = (bf16_t*)carve((size_t)LAYERS * HF * HF * 2);
  bf16_t* f_px1 = (bf16_t*)carve((size_t)LAYERS * HF * HF * 2);
  bf16_t* f_ph1 = (bf16_t*)carve((size_t)LAYERS * 256 * HF * 2);
  bf16_t* f_ph2 = (bf16_t*)carve((size_t)LAYERS * HF * HF * 2);
  (void)ws_size; (void)n_in; (void)out_size;

  // ---- degrees, working x, embedding ----
  hipMemsetAsync(cdeg, 0, (size_t)N * 4, stream);
  egnn_degree<<<(E + 255) / 256, 256, 0, stream>>>(src, cdeg, E);
  hipMemcpyAsync(xw, x_in, (size_t)N * 3 * 4, hipMemcpyDeviceToDevice, stream);
  egnn_emb<<<((size_t)N * HF + 255) / 256, 256, 0, stream>>>(h_in, emb_w, emb_b,
                                                             h32, hbf, N);
  // ---- weight fragment conversion (bf16, WMMA B layout) ----
  for (int l = 0; l < LAYERS; ++l) {
    egnn_wfrag<<<(KPAD1 * HF + 255) / 256, 256, 0, stream>>>(
        pe_w1 + (size_t)l * DIN * HF, f_pe1 + (size_t)l * KPAD1 * HF, DIN, KPAD1);
    egnn_wfrag<<<(HF * HF + 255) / 256, 256, 0, stream>>>(
        pe_w2 + (size_t)l * HF * HF, f_pe2 + (size_t)l * HF * HF, HF, HF);
    egnn_wfrag<<<(HF * HF + 255) / 256, 256, 0, stream>>>(
        px_w1 + (size_t)l * HF * HF, f_px1 + (size_t)l * HF * HF, HF, HF);
    egnn_wfrag<<<(256 * HF + 255) / 256, 256, 0, stream>>>(
        ph_w1 + (size_t)l * 256 * HF, f_ph1 + (size_t)l * 256 * HF, 256, 256);
    egnn_wfrag<<<(HF * HF + 255) / 256, 256, 0, stream>>>(
        ph_w2 + (size_t)l * HF * HF, f_ph2 + (size_t)l * HF * HF, HF, HF);
  }

  const int etiles = (E + 15) / 16;
  const int eblocks = (etiles + 7) / 8;
  const int ntiles = (N + 15) / 16;
  const int nblocks = (ntiles + 7) / 8;

  for (int l = 0; l < LAYERS; ++l) {
    hipMemsetAsync(aggh, 0, (size_t)N * HF * 4, stream);
    hipMemsetAsync(aggx, 0, (size_t)N * 3 * 4, stream);
    egnn_edge<<<eblocks, 256, 0, stream>>>(
        src, dst, xw, eattr, hbf,
        f_pe1 + (size_t)l * KPAD1 * HF, pe_b1 + l * HF, pe_g1 + l * HF,
        pe_bb1 + l * HF,
        f_pe2 + (size_t)l * HF * HF, pe_b2 + l * HF, pe_g2 + l * HF,
        pe_bb2 + l * HF,
        f_px1 + (size_t)l * HF * HF, px_b1 + l * HF, px_g1 + l * HF,
        px_bb1 + l * HF,
        px_w2 + (size_t)l * HF, px_b2 + l, aggh, aggx, E);
    egnn_node<<<nblocks, 256, 0, stream>>>(
        aggh, aggx, cdeg, h32, hbf, xw,
        f_ph1 + (size_t)l * 256 * HF, ph_b1 + l * HF, ph_g1 + l * HF,
        ph_bb1 + l * HF,
        f_ph2 + (size_t)l * HF * HF, ph_b2 + l * HF, N);
  }

  egnn_final<<<((size_t)N * 19 + 255) / 256, 256, 0, stream>>>(
      h32, xw, out_w, out_b, (float*)d_out, N);
}
